// GIN_8950711845676
// MI455X (gfx1250) — matrix operand order repaired and checked by reading, not compiled, not run on hardware
//
#include <hip/hip_runtime.h>
#include <hip/hip_bf16.h>
#include <math.h>

typedef __bf16 bf16_t;
typedef __attribute__((ext_vector_type(16))) __bf16 v16bf;
typedef __attribute__((ext_vector_type(8)))  float  v8f;

#define HID 64
#define NWAVES 8   // waves per block in the WMMA MLP kernel

// ---------------------------------------------------------------------------
// fp32 -> bf16 round-to-nearest-even (for weight packing)
static __device__ __forceinline__ unsigned short f2bf(float f) {
  union { float f; unsigned u; } v; v.f = f;
  unsigned r = v.u + 0x7fffu + ((v.u >> 16) & 1u);
  return (unsigned short)(r >> 16);
}

// ---------------------------------------------------------------------------
// Pack 8 weight matrices [Kact x 64] fp32 into WMMA bf16 B-fragment layout.
// Fragment layout per matrix: frag = n*2 + kc  (n = 16-col tile 0..3, kc = K
// chunk 0..1).  Within a fragment: lane L, elem e ->
//   N = n*16 + (L&15),  K = kc*32 + ((L>>4)&1)*8 + (e<8 ? e : e+8)  (ISA 05_wmma B)
// Storage: dst[mat*4096 + frag*512 + lane*16 + e]  (contiguous 32B per lane)
struct WPrepArgs {
  const float* src[8];
  int kact[8];
  unsigned short* dst;
};

__global__ void k_prep_weights(WPrepArgs p) {
  int gid = blockIdx.x * blockDim.x + threadIdx.x;
  if (gid >= 8 * 4096) return;
  int mat  = gid >> 12;
  int idx  = gid & 4095;
  int frag = idx >> 9;
  int lane = (idx >> 4) & 31;
  int e    = idx & 15;
  int n  = frag >> 1, kc = frag & 1;
  int K = kc * 32 + ((lane >> 4) & 1) * 8 + (e < 8 ? e : e + 8);
  int N = n * 16 + (lane & 15);
  float v = (K < p.kact[mat]) ? p.src[mat][K * 64 + N] : 0.f;
  p.dst[mat * 4096 + idx] = f2bf(v);
}

// ---------------------------------------------------------------------------
// Edge message + scatter:  agg[dst] += relu(h[src] + ea@ew + eb)
// One wave per edge; lane covers features 2*lane, 2*lane+1.  Atomics to the
// 12.8MB agg array stay resident in L2.  Relu-zero messages skip the atomic.
template <int D>
__global__ void k_edge_msg(const float* __restrict__ hsrc,
                           const int*   __restrict__ ei,
                           const float* __restrict__ ea,
                           const float* __restrict__ ew,
                           const float* __restrict__ eb,
                           float* __restrict__ agg, int E) {
  int wid  = (int)((blockIdx.x * blockDim.x + threadIdx.x) >> 5);
  int lane = threadIdx.x & 31;
  if (wid >= E) return;
  int src = ei[wid];
  int dst = ei[E + wid];
  float eav = (lane < 3) ? ea[wid * 3 + lane] : 0.f;
  float ea0 = __shfl(eav, 0);
  float ea1 = __shfl(eav, 1);
  float ea2 = __shfl(eav, 2);
  int f0 = lane * 2;
#pragma unroll
  for (int j = 0; j < 2; ++j) {
    int f = f0 + j;
    if (f < D) {
      float m = hsrc[src * D + f]
              + ea0 * ew[f] + ea1 * ew[D + f] + ea2 * ew[2 * D + f] + eb[f];
      if (m > 0.f) atomicAdd(&agg[dst * HID + f], m);
    }
  }
}

// ---------------------------------------------------------------------------
// Fused 2-layer MLP on WMMA matrix cores.
//   out = epi( relu( (inA(+inB)) @ W1 + b1 ) @ W2 + b2 )
// epi: optional relu2, optional h*gamma*bnscale + beta (BatchNorm eval).
// One wave per 16-row tile; 16x v_wmma_f32_16x16x32_bf16 per wave.
// Inter-GEMM C->A relayout staged through wave-private LDS.
__global__ void __launch_bounds__(256)
k_mlp_wmma(const float* __restrict__ inA, int lda, int da,
           const float* __restrict__ inB,                     // optional [rows,64]
           const unsigned short* __restrict__ w1f, const float* __restrict__ b1,
           const unsigned short* __restrict__ w2f, const float* __restrict__ b2,
           const float* __restrict__ gamma, const float* __restrict__ beta,
           float bnscale, int relu2,
           float* __restrict__ out, int rows) {
  __shared__ bf16_t lds[NWAVES][16 * HID];
  int wv   = threadIdx.x >> 5;
  int lane = threadIdx.x & 31;
  int tile = blockIdx.x * NWAVES + wv;
  int r0   = tile * 16;
  int arow = r0 + (lane & 15);
  bool aok = (arow < rows);
  int koff = (lane >> 4) ? 8 : 0;
  int coln = lane & 15;

  // ---- A fragments for GEMM1 (ISA 16-bit A layout) ----
  v16bf A[2];
#pragma unroll
  for (int kc = 0; kc < 2; ++kc) {
#pragma unroll
    for (int e = 0; e < 16; ++e) {
      int K = kc * 32 + koff + (e < 8 ? e : e + 8);
      float v = 0.f;
      if (aok) {
        if (K < da) v = inA[arow * lda + K];
        if (inB)    v += inB[arow * HID + K];
      }
      A[kc][e] = (bf16_t)v;
    }
  }

  // ---- GEMM1: 4 col-tiles x 2 K-chunks, bias + relu -> LDS (bf16) ----
#pragma unroll
  for (int n = 0; n < 4; ++n) {
    v8f c = {};
    v16bf wb0 = *(const v16bf*)(w1f + (n * 2 + 0) * 512 + lane * 16);
    v16bf wb1 = *(const v16bf*)(w1f + (n * 2 + 1) * 512 + lane * 16);
    c = __builtin_amdgcn_wmma_f32_16x16x32_bf16(false, A[0], false, wb0,
                                                (short)0, c, false, false);
    c = __builtin_amdgcn_wmma_f32_16x16x32_bf16(false, A[1], false, wb1,
                                                (short)0, c, false, false);
    int col = n * 16 + coln;
    float bias = b1[col];
#pragma unroll
    for (int r = 0; r < 8; ++r) {
      int m = ((lane >> 4) & 1) * 8 + r;            // C layout: VGPR r -> row
      float h = c[r] + bias;
      h = h > 0.f ? h : 0.f;
      lds[wv][m * HID + col] = (bf16_t)h;
    }
  }
  __syncthreads();

  // ---- Relayout: read back as A fragments for GEMM2 ----
  v16bf A2[2];
#pragma unroll
  for (int kc = 0; kc < 2; ++kc) {
#pragma unroll
    for (int e = 0; e < 16; ++e) {
      int K = kc * 32 + koff + (e < 8 ? e : e + 8);
      A2[kc][e] = lds[wv][(lane & 15) * HID + K];
    }
  }

  // ---- GEMM2 + epilogue ----
#pragma unroll
  for (int n = 0; n < 4; ++n) {
    v8f c = {};
    v16bf wb0 = *(const v16bf*)(w2f + (n * 2 + 0) * 512 + lane * 16);
    v16bf wb1 = *(const v16bf*)(w2f + (n * 2 + 1) * 512 + lane * 16);
    c = __builtin_amdgcn_wmma_f32_16x16x32_bf16(false, A2[0], false, wb0,
                                                (short)0, c, false, false);
    c = __builtin_amdgcn_wmma_f32_16x16x32_bf16(false, A2[1], false, wb1,
                                                (short)0, c, false, false);
    int col = n * 16 + coln;
    float bias = b2[col];
    float sc = 1.f, bt = 0.f;
    if (gamma) { sc = gamma[col] * bnscale; bt = beta[col]; }
#pragma unroll
    for (int r = 0; r < 8; ++r) {
      int m = ((lane >> 4) & 1) * 8 + r;
      int orow = r0 + m;
      if (orow < rows) {
        float h = c[r] + bias;
        if (relu2) h = h > 0.f ? h : 0.f;
        out[orow * HID + col] = h * sc + bt;
      }
    }
  }
}

// ---------------------------------------------------------------------------
// Global mean pool: wave per node, lane covers 2 features (coalesced float2).
__global__ void k_pool(const float* __restrict__ h, const int* __restrict__ batch,
                       float* __restrict__ sums, float* __restrict__ cnts, int n) {
  int wid  = (int)((blockIdx.x * blockDim.x + threadIdx.x) >> 5);
  int lane = threadIdx.x & 31;
  if (wid >= n) return;
  int g = batch[wid];
  float2 hv = *(const float2*)(h + wid * HID + lane * 2);
  atomicAdd(&sums[g * HID + lane * 2 + 0], hv.x);
  atomicAdd(&sums[g * HID + lane * 2 + 1], hv.y);
  if (lane == 0) atomicAdd(&cnts[g], 1.f);
}

__global__ void k_pool_div(const float* __restrict__ sums,
                           const float* __restrict__ cnts,
                           float* __restrict__ pooled, int total) {
  int gid = blockIdx.x * blockDim.x + threadIdx.x;
  if (gid >= total) return;
  float c = cnts[gid >> 6];
  c = c > 1.f ? c : 1.f;
  pooled[gid] = sums[gid] / c;
}

// ---------------------------------------------------------------------------
extern "C" void kernel_launch(void* const* d_in, const int* in_sizes, int n_in,
                              void* d_out, int out_size, void* d_ws, size_t ws_size,
                              hipStream_t stream) {
  (void)in_sizes; (void)n_in; (void)out_size; (void)ws_size;
  const int N = 50000, E = 1600000, G = 1024;
  const float BN_SCALE = 1.0f / sqrtf(1.0f + 1e-5f);

  const float* x     = (const float*)d_in[0];
  const int*   ei    = (const int*)d_in[1];
  const float* ea    = (const float*)d_in[2];
  const int*   batch = (const int*)d_in[3];
  // params flattened in dict order: per layer l: ew,eb,w1,b1,w2,b2,g,be
  const float* ew[3], *eb[3], *w1[3], *b1[3], *w2[3], *b2[3], *gm[3], *be[3];
  for (int l = 0; l < 3; ++l) {
    const int b = 4 + l * 8;
    ew[l] = (const float*)d_in[b + 0]; eb[l] = (const float*)d_in[b + 1];
    w1[l] = (const float*)d_in[b + 2]; b1[l] = (const float*)d_in[b + 3];
    w2[l] = (const float*)d_in[b + 4]; b2[l] = (const float*)d_in[b + 5];
    gm[l] = (const float*)d_in[b + 6]; be[l] = (const float*)d_in[b + 7];
  }
  const float* lw1 = (const float*)d_in[28];
  const float* lb1 = (const float*)d_in[29];
  const float* lw2 = (const float*)d_in[30];
  const float* lb2 = (const float*)d_in[31];

  // workspace carve-up (256B aligned)
  char* ws = (char*)d_ws;
  size_t off = 0;
  auto carve = [&](size_t bytes) -> void* {
    void* p = ws + off;
    off = (off + bytes + 255) & ~(size_t)255;
    return p;
  };
  float* hA     = (float*)carve((size_t)N * HID * 4);
  float* hB     = (float*)carve((size_t)N * HID * 4);
  float* agg    = (float*)carve((size_t)N * HID * 4);
  float* sums   = (float*)carve((size_t)G * HID * 4);
  float* cnts   = (float*)carve((size_t)G * 4);
  float* pooled = (float*)carve((size_t)G * HID * 4);
  unsigned short* wfr = (unsigned short*)carve((size_t)8 * 4096 * 2);

  // 1) pack all weight matrices into WMMA bf16 B-fragments
  WPrepArgs pa;
  const float* srcs[8] = { w1[0], w2[0], w1[1], w2[1], w1[2], w2[2], lw1, lw2 };
  int kacts[8]         = { 9, 64, 64, 64, 64, 64, 64, 64 };
  for (int i = 0; i < 8; ++i) { pa.src[i] = srcs[i]; pa.kact[i] = kacts[i]; }
  pa.dst = wfr;
  k_prep_weights<<<(8 * 4096 + 255) / 256, 256, 0, stream>>>(pa);

  const int edgeBlocks = (int)(((size_t)E * 32 + 255) / 256);
  const int mlpBlocksN = ((N + 15) / 16 + NWAVES - 1) / NWAVES;

  // 2) layer 0 (input dim 9)
  hipMemsetAsync(agg, 0, (size_t)N * HID * 4, stream);
  k_edge_msg<9><<<edgeBlocks, 256, 0, stream>>>(x, ei, ea, ew[0], eb[0], agg, E);
  k_mlp_wmma<<<mlpBlocksN, 256, 0, stream>>>(x, 9, 9, agg,
      wfr + 0 * 4096, b1[0], wfr + 1 * 4096, b2[0],
      gm[0], be[0], BN_SCALE, 1, hA, N);

  // 3) layer 1
  hipMemsetAsync(agg, 0, (size_t)N * HID * 4, stream);
  k_edge_msg<64><<<edgeBlocks, 256, 0, stream>>>(hA, ei, ea, ew[1], eb[1], agg, E);
  k_mlp_wmma<<<mlpBlocksN, 256, 0, stream>>>(hA, HID, HID, agg,
      wfr + 2 * 4096, b1[1], wfr + 3 * 4096, b2[1],
      gm[1], be[1], BN_SCALE, 1, hB, N);

  // 4) layer 2
  hipMemsetAsync(agg, 0, (size_t)N * HID * 4, stream);
  k_edge_msg<64><<<edgeBlocks, 256, 0, stream>>>(hB, ei, ea, ew[2], eb[2], agg, E);
  k_mlp_wmma<<<mlpBlocksN, 256, 0, stream>>>(hB, HID, HID, agg,
      wfr + 4 * 4096, b1[2], wfr + 5 * 4096, b2[2],
      gm[2], be[2], BN_SCALE, 1, hA, N);

  // 5) global mean pool
  hipMemsetAsync(sums, 0, (size_t)G * HID * 4, stream);
  hipMemsetAsync(cnts, 0, (size_t)G * 4, stream);
  k_pool<<<(int)(((size_t)N * 32 + 255) / 256), 256, 0, stream>>>(hA, batch, sums, cnts, N);
  k_pool_div<<<(G * HID + 255) / 256, 256, 0, stream>>>(sums, cnts, pooled, G * HID);

  // 6) head MLP: relu(pooled@lw1+lb1)@lw2+lb2  (no relu2, no BN)
  const int mlpBlocksG = ((G + 15) / 16 + NWAVES - 1) / NWAVES;
  k_mlp_wmma<<<mlpBlocksG, 256, 0, stream>>>(pooled, HID, HID, nullptr,
      wfr + 6 * 4096, lb1, wfr + 7 * 4096, lb2,
      nullptr, nullptr, 1.0f, 0, (float*)d_out, G);
}
